// AttentionLayer_32779190403782
// MI455X (gfx1250) — matrix operand order repaired and checked by reading, not compiled, not run on hardware
//
#include <hip/hip_runtime.h>

#define BB   4
#define CC   256
#define NN   4096
#define CQKD 32
#define JT   32
#define VST  40            // LDS V-tile row stride in halfs ([c][j], 80B rows, 16B aligned)
#define SSTR 33            // S rows stride (floats)
#define PSTR 40            // P rows stride (halfs), keeps 16B-aligned fragment chunks
#define OSTR 19            // epilogue transpose stride (floats), conflict-free both lane halves

typedef __attribute__((ext_vector_type(16))) _Float16 v16h;
typedef __attribute__((ext_vector_type(8)))  float    v8f;
typedef __attribute__((ext_vector_type(4)))  int      v4i;

#if __has_builtin(__builtin_amdgcn_global_load_async_to_lds_b128) && \
    __has_builtin(__builtin_amdgcn_s_wait_asynccnt)
#define HAVE_ASYNC_LDS 1
typedef __attribute__((address_space(1))) v4i as1_v4i;   // global
typedef __attribute__((address_space(3))) v4i as3_v4i;   // LDS
#endif

// ---------------------------------------------------------------------------
// Kernel 1: fused 1x1-conv + BatchNorm (eval) projection -> f16 q, k, v.
// q,k: [B][N][32] (row-major per position, matches WMMA A/B fragment chunks),
// v  : [B][C][N] (channel-major: coalesced stores here AND contiguous k-pairs
//      for the P*V WMMA B-fragments later).
// ---------------------------------------------------------------------------
__launch_bounds__(256)
__global__ void qkv_proj_kernel(const float* __restrict__ x,
    const float* __restrict__ qw, const float* __restrict__ qb, const float* __restrict__ qg,
    const float* __restrict__ qbe, const float* __restrict__ qmn, const float* __restrict__ qvr,
    const float* __restrict__ kw, const float* __restrict__ kb, const float* __restrict__ kg,
    const float* __restrict__ kbe, const float* __restrict__ kmn, const float* __restrict__ kvr,
    const float* __restrict__ vw, const float* __restrict__ vb, const float* __restrict__ vg,
    const float* __restrict__ vbe, const float* __restrict__ vmn, const float* __restrict__ vvr,
    _Float16* __restrict__ Qo, _Float16* __restrict__ Ko, _Float16* __restrict__ Vo)
{
    const int t   = blockIdx.x * blockDim.x + threadIdx.x;   // 0 .. 320*B*N-1
    const int og  = t >> 14;                                 // fused out-channel 0..319
    const int pos = t & 16383;                               // b*N + n
    const int b   = pos >> 12;
    const int n   = pos & 4095;

    const float *w, *bi, *gg, *be, *mn, *vr;
    _Float16* dst;
    int o;
    if (og < 32) {
        o = og;       w = qw; bi = qb; gg = qg; be = qbe; mn = qmn; vr = qvr;
        dst = Qo + (size_t)pos * CQKD + o;
    } else if (og < 64) {
        o = og - 32;  w = kw; bi = kb; gg = kg; be = kbe; mn = kmn; vr = kvr;
        dst = Ko + (size_t)pos * CQKD + o;
    } else {
        o = og - 64;  w = vw; bi = vb; gg = vg; be = vbe; mn = vmn; vr = vvr;
        dst = Vo + ((size_t)b * CC + o) * NN + n;            // channel-major, coalesced
    }

    const float* wrow = w + (size_t)o * CC;
    const float* xcol = x + ((size_t)b * CC) * NN + n;       // stride NN over channels
    float acc = 0.f;
#pragma unroll 8
    for (int c = 0; c < CC; ++c)
        acc += wrow[c] * xcol[(size_t)c << 12];

    const float sc = gg[o] * rsqrtf(vr[o] + 1e-5f);
    const float y  = (acc + bi[o]) * sc + (be[o] - mn[o] * sc);
    *dst = (_Float16)y;
}

// ---------------------------------------------------------------------------
// Kernel 2: fused flash-attention + gamma*out + residual.
// Grid: B*(N/16) blocks, 128 threads (4 waves). One 16-row i-tile per block.
// Wave 0: S = Q K^T via v_wmma_f32_16x16x32_f16 (K-dim == 32 channels),
//         online softmax through LDS; all waves: O += P V (4 WMMAs / wave,
//         each wave owns 64 of 256 channels). Tiles staged to LDS with
//         global_load_async_to_lds_b128; all fragments are ds_load_b128 pairs.
// ---------------------------------------------------------------------------
__launch_bounds__(128, 1)
__global__ void attn_kernel(const _Float16* __restrict__ Q,
                            const _Float16* __restrict__ K,
                            const _Float16* __restrict__ V,
                            const float*    __restrict__ X,
                            const float*    __restrict__ gamma,
                            float*          __restrict__ Out)
{
    __shared__ __align__(16) _Float16 Kt[JT * CQKD];     // [j][c]   2 KB
    __shared__ __align__(16) _Float16 Vt[CC * VST];      // [c][j]  20 KB
    __shared__ __align__(16) float     Sb[16 * SSTR];    // logits 16 x 32
    __shared__ __align__(16) _Float16  Pb[16 * PSTR];    // probs  16 x 32 (f16)
    __shared__ __align__(16) float     OutT[CC * OSTR];  // epilogue transpose ~19.5 KB
    __shared__ float Alpha[16];
    __shared__ float Linv[16];

    const int tid  = threadIdx.x;
    const int lane = tid & 31;
    const int wave = tid >> 5;
    const int b    = blockIdx.x >> 8;            // 256 i-tiles per batch
    const int i0   = (blockIdx.x & 255) << 4;

    const int mrow = lane & 15;                  // A: row M / B,C,D: col N
    const int kh   = lane >> 4;                  // lane-half selector

    // ---- Q fragment, A-layout 16x32 f16: two contiguous 16B chunks ----
    v16h qfrag;
    {
        const _Float16* qb_ = Q + ((size_t)b * NN + i0 + mrow) * CQKD + kh * 16;
        ((uint4*)&qfrag)[0] = *(const uint4*)(qb_);
        ((uint4*)&qfrag)[1] = *(const uint4*)(qb_ + 8);
    }

    v8f acc[4] = {};                 // 16 rows x 64 channels per wave, f32
    float mrun = -1e30f;             // per-row softmax state (lanes 0-15, wave 0)
    float lrun = 0.f;

    const _Float16* kbase = K + ((size_t)b * NN) * CQKD;
    const _Float16* vbase = V + ((size_t)b * CC) * NN;

    for (int j0 = 0; j0 < NN; j0 += JT) {
        __syncthreads();             // previous iteration's LDS consumers done

        // ---- stage K tile [j][c] (128 x b128) and V tile [c][j] (8 x b128/thr) ----
        const _Float16* ks = kbase + (size_t)j0 * CQKD;       // 32x32 contiguous
#ifdef HAVE_ASYNC_LDS
        __builtin_amdgcn_global_load_async_to_lds_b128(
            (as1_v4i*)((const uint4*)ks + tid),
            (as3_v4i*)((uint4*)Kt + tid), 0, 0);
#pragma unroll
        for (int rep = 0; rep < 8; ++rep) {
            const int idx  = rep * 128 + tid;     // 0..1023
            const int c    = idx >> 2;            // channel row
            const int part = idx & 3;             // 16B chunk within 64B row
            __builtin_amdgcn_global_load_async_to_lds_b128(
                (as1_v4i*)((const uint4*)(vbase + (size_t)c * NN + j0) + part),
                (as3_v4i*)((uint4*)(Vt + c * VST) + part), 0, 0);
        }
#else
        ((uint4*)Kt)[tid] = ((const uint4*)ks)[tid];
#pragma unroll
        for (int rep = 0; rep < 8; ++rep) {
            const int idx  = rep * 128 + tid;
            const int c    = idx >> 2;
            const int part = idx & 3;
            ((uint4*)(Vt + c * VST))[part] =
                ((const uint4*)(vbase + (size_t)c * NN + j0))[part];
        }
#endif
        if (j0 + JT < NN) {          // prefetch next tiles -> global_prefetch_b8
            __builtin_prefetch(kbase + (size_t)(j0 + JT) * CQKD + (tid << 3), 0, 3);
            __builtin_prefetch(vbase + (size_t)(tid * 2 + 0) * NN + j0 + JT, 0, 3);
            __builtin_prefetch(vbase + (size_t)(tid * 2 + 1) * NN + j0 + JT, 0, 3);
        }
#ifdef HAVE_ASYNC_LDS
        __builtin_amdgcn_s_wait_asynccnt(0);
#endif
        __syncthreads();

        if (wave == 0) {
            // ---- S = Q K^T : two 16x16x32 WMMAs (j halves) ----
            // B-layout: lane half kh covers K (channels) kh*16..kh*16+15, contiguous
            v16h kf0, kf1;
            {
                const _Float16* p0 = Kt + mrow * CQKD + kh * 16;
                const _Float16* p1 = Kt + (mrow + 16) * CQKD + kh * 16;
                ((uint4*)&kf0)[0] = *(const uint4*)(p0);
                ((uint4*)&kf0)[1] = *(const uint4*)(p0 + 8);
                ((uint4*)&kf1)[0] = *(const uint4*)(p1);
                ((uint4*)&kf1)[1] = *(const uint4*)(p1 + 8);
            }
            v8f s0 = {}, s1 = {};
            s0 = __builtin_amdgcn_wmma_f32_16x16x32_f16(false, qfrag, false, kf0,
                                                        (short)0, s0, false, false);
            s1 = __builtin_amdgcn_wmma_f32_16x16x32_f16(false, qfrag, false, kf1,
                                                        (short)0, s1, false, false);
            // scatter S to LDS (C-layout: vgpr r -> rows r / r+8)
            const int roff = kh * 8;
#pragma unroll
            for (int r = 0; r < 8; ++r) {
                Sb[(r + roff) * SSTR + mrow]      = s0[r];
                Sb[(r + roff) * SSTR + 16 + mrow] = s1[r];
            }
            __asm__ volatile("s_wait_dscnt 0" ::: "memory");

            if (tid < 16) {          // one lane per i-row: online softmax
                float mv = mrun;
#pragma unroll
                for (int j = 0; j < JT; ++j)
                    mv = fmaxf(mv, Sb[tid * SSTR + j]);
                const float al = __expf(mrun - mv);
                float sum = 0.f;
#pragma unroll
                for (int j = 0; j < JT; ++j) {
                    const float p = __expf(Sb[tid * SSTR + j] - mv);
                    sum += p;
                    Pb[tid * PSTR + j] = (_Float16)p;
                }
                lrun = al * lrun + sum;
                mrun = mv;
                Alpha[tid] = al;
            }
            __asm__ volatile("s_wait_dscnt 0" ::: "memory");
        }
        __syncthreads();             // publish Pb / Alpha to all waves

        // ---- rescale accumulators by alpha(row) ----
        const int roff = kh * 8;
        float av[8];
#pragma unroll
        for (int r = 0; r < 8; ++r) av[r] = Alpha[r + roff];
#pragma unroll
        for (int r = 0; r < 8; ++r) {
            acc[0][r] *= av[r]; acc[1][r] *= av[r];
            acc[2][r] *= av[r]; acc[3][r] *= av[r];
        }

        // ---- P fragment, A-layout: two contiguous 16B chunks ----
        v16h pfrag;
        {
            const _Float16* pp = Pb + mrow * PSTR + kh * 16;
            ((uint4*)&pfrag)[0] = *(const uint4*)(pp);
            ((uint4*)&pfrag)[1] = *(const uint4*)(pp + 8);
        }

        // ---- O += P * V : 4 WMMAs, 16 channels each ----
        // B-layout: lane half kh covers K (tile-j) kh*16..kh*16+15, contiguous
        const int cbase = wave * 64 + mrow;
#pragma unroll
        for (int t = 0; t < 4; ++t) {
            v16h vf;
            const _Float16* vp = Vt + (cbase + t * 16) * VST + kh * 16;
            ((uint4*)&vf)[0] = *(const uint4*)(vp);
            ((uint4*)&vf)[1] = *(const uint4*)(vp + 8);
            acc[t] = __builtin_amdgcn_wmma_f32_16x16x32_f16(false, pfrag, false, vf,
                                                            (short)0, acc[t], false, false);
        }
    }

    // ---- epilogue: O/l * gamma via LDS transpose, then coalesced +residual ----
    if (wave == 0 && tid < 16) Linv[tid] = 1.0f / lrun;
    __syncthreads();

    const float g = gamma[0];
    const int roff = kh * 8;
    float lin[8];
#pragma unroll
    for (int r = 0; r < 8; ++r) lin[r] = Linv[r + roff];

#pragma unroll
    for (int t = 0; t < 4; ++t) {
        const int c = wave * 64 + t * 16 + mrow;
#pragma unroll
        for (int r = 0; r < 8; ++r)
            OutT[c * OSTR + (r + roff)] = g * acc[t][r] * lin[r];
    }
    __syncthreads();

    // each thread writes two 64B channel rows: fully coalesced global traffic
#pragma unroll
    for (int rr = 0; rr < 2; ++rr) {
        const int c = tid * 2 + rr;
        const size_t gb = (((size_t)b * CC + c) << 12) + i0;
        const float4* xrow = (const float4*)(X + gb);
        float4*       orow = (float4*)(Out + gb);
#pragma unroll
        for (int q = 0; q < 4; ++q) {
            const float4 xv = xrow[q];
            float4 ov;
            ov.x = OutT[c * OSTR + 4 * q + 0] + xv.x;
            ov.y = OutT[c * OSTR + 4 * q + 1] + xv.y;
            ov.z = OutT[c * OSTR + 4 * q + 2] + xv.z;
            ov.w = OutT[c * OSTR + 4 * q + 3] + xv.w;
            orow[q] = ov;
        }
    }
}

// ---------------------------------------------------------------------------
extern "C" void kernel_launch(void* const* d_in, const int* in_sizes, int n_in,
                              void* d_out, int out_size, void* d_ws, size_t ws_size,
                              hipStream_t stream)
{
    (void)in_sizes; (void)n_in; (void)out_size; (void)ws_size;

    const float* x   = (const float*)d_in[0];
    const float* qw  = (const float*)d_in[1];
    const float* qb  = (const float*)d_in[2];
    const float* qg  = (const float*)d_in[3];
    const float* qbe = (const float*)d_in[4];
    const float* qmn = (const float*)d_in[5];
    const float* qvr = (const float*)d_in[6];
    const float* kw  = (const float*)d_in[7];
    const float* kb  = (const float*)d_in[8];
    const float* kg  = (const float*)d_in[9];
    const float* kbe = (const float*)d_in[10];
    const float* kmn = (const float*)d_in[11];
    const float* kvr = (const float*)d_in[12];
    const float* vw  = (const float*)d_in[13];
    const float* vb  = (const float*)d_in[14];
    const float* vg  = (const float*)d_in[15];
    const float* vbe = (const float*)d_in[16];
    const float* vmn = (const float*)d_in[17];
    const float* vvr = (const float*)d_in[18];
    const float* gam = (const float*)d_in[19];

    _Float16* Qo = (_Float16*)d_ws;                         // 1 MB
    _Float16* Ko = Qo + (size_t)BB * NN * CQKD;             // 1 MB
    _Float16* Vo = Ko + (size_t)BB * NN * CQKD;             // 8 MB

    const int projThreads = 320 * BB * NN;                  // 5,242,880
    qkv_proj_kernel<<<projThreads / 256, 256, 0, stream>>>(
        x, qw, qb, qg, qbe, qmn, qvr,
           kw, kb, kg, kbe, kmn, kvr,
           vw, vb, vg, vbe, vmn, vvr,
        Qo, Ko, Vo);

    attn_kernel<<<BB * (NN / 16), 128, 0, stream>>>(
        Qo, Ko, Vo, x, gam, (float*)d_out);
}
